// MultiAgentActorCritic_46076409152038
// MI455X (gfx1250) — matrix operand order repaired and checked by reading, not compiled, not run on hardware
//
#include <hip/hip_runtime.h>
#include <hip/hip_bf16.h>

// ---------------------------------------------------------------------------
// MultiAgentActorCritic forward for MI455X (gfx1250, wave32, WMMA f16).
// All heavy GEMMs (convs via implicit im2col, frontier MLP chain, QK scores)
// run on v_wmma_f32_16x16x32_f16. BN folded into per-channel scale/shift.
// ---------------------------------------------------------------------------

typedef __attribute__((ext_vector_type(16))) _Float16 v16h;
typedef __attribute__((ext_vector_type(8)))  float    v8f;

#define MAXF 8192

__device__ __forceinline__ int imin(int a, int b) { return a < b ? a : b; }
__device__ __forceinline__ int imax(int a, int b) { return a > b ? a : b; }

__device__ __forceinline__ v8f wmma_f16(v16h a, v16h b, v8f c) {
    return __builtin_amdgcn_wmma_f32_16x16x32_f16(false, a, false, b, (short)0, c, false, false);
}

// A-fragment (16 rows x 32 K, f16) from an LDS tile with row-stride `ld`.
// Layout: lanes 0-15 hold M=lane, K = {k0..k0+15};
//         lanes 16-31 hold M=lane-16, K = {k0+16..k0+31}.
__device__ __forceinline__ v16h load_a_frag(const _Float16* base, int ld, int m0, int k0) {
    int lane = threadIdx.x & 31;
    const _Float16* row = base + (size_t)(m0 + (lane & 15)) * ld + k0 + ((lane >> 4) << 4);
    v16h a;
#pragma unroll
    for (int e = 0; e < 16; ++e) { a[e] = row[e]; }
    return a;
}

// B-fragment: packed so lane reads 16 contiguous halves (32B -> 2x b128).
__device__ __forceinline__ v16h load_b_frag(const _Float16* p) {
    int lane = threadIdx.x & 31;
    return *(const v16h*)(p + (size_t)lane * 16);
}

// ---------------------------------------------------------------------------
// Param packing / BN folding
// ---------------------------------------------------------------------------

// scale = g/sqrt(v+eps), shift = (conv_b - m)*scale + bn_b
__global__ void fold_bn_kernel(const float* cb, const float* g, const float* bb,
                               const float* m, const float* v,
                               float* scale, float* shift, int C) {
    int c = blockIdx.x * blockDim.x + threadIdx.x;
    if (c < C) {
        float inv = g[c] / sqrtf(v[c] + 1e-5f);
        scale[c] = inv;
        shift[c] = (cb[c] - m[c]) * inv + bb[c];
    }
}

__global__ void ident_scale_kernel(const float* cb, float* scale, float* shift, int C) {
    int c = blockIdx.x * blockDim.x + threadIdx.x;
    if (c < C) { scale[c] = 1.f; shift[c] = cb[c]; }
}

// Conv weights OIHW (COUT,CIN,3,3) -> packed B fragments.
// K-index: k = tap*CIN + cin  (tap = ky*3+kx), padded to 32 with zeros.
// packed[nt][ks][lane][e], lane l: n = nt*16 + (l&15), k = ks*32 + (l<16?0:16) + e.
__global__ void pack_conv_w_kernel(const float* __restrict__ w, _Float16* __restrict__ out,
                                   int CIN, int COUT, int KSTEPS) {
    int KTOT = 9 * CIN;
    int total = (COUT / 16) * KSTEPS * 512;
    for (int t = blockIdx.x * blockDim.x + threadIdx.x; t < total; t += gridDim.x * blockDim.x) {
        int e = t & 15, lane = (t >> 4) & 31;
        int ks = (t >> 9) % KSTEPS, nt = (t >> 9) / KSTEPS;
        int n = nt * 16 + (lane & 15);
        int k = ks * 32 + ((lane >> 4) << 4) + e;
        float v = 0.f;
        if (k < KTOT) {
            int tap = k / CIN, cin = k % CIN;
            v = w[((size_t)n * CIN + cin) * 9 + tap];
        }
        out[t] = (_Float16)v;
    }
}

// Linear weights (IN,OUT) row-major (x @ W) -> packed B fragments.
__global__ void pack_lin_w_kernel(const float* __restrict__ w, _Float16* __restrict__ out,
                                  int IN, int OUT, int KSTEPS) {
    int total = (OUT / 16) * KSTEPS * 512;
    for (int t = blockIdx.x * blockDim.x + threadIdx.x; t < total; t += gridDim.x * blockDim.x) {
        int e = t & 15, lane = (t >> 4) & 31;
        int ks = (t >> 9) % KSTEPS, nt = (t >> 9) / KSTEPS;
        int n = nt * 16 + (lane & 15);
        int k = ks * 32 + ((lane >> 4) << 4) + e;
        out[t] = (_Float16)((k < IN) ? w[(size_t)k * OUT + n] : 0.f);
    }
}

// Q (B,8,128) f32 -> per-batch B fragments (N padded 8->16 with zeros).
__global__ void pack_q_kernel(const float* __restrict__ Qf, _Float16* __restrict__ qp) {
    int t = blockIdx.x * blockDim.x + threadIdx.x;
    if (t >= 32 * 4 * 512) return;
    int e = t & 15, lane = (t >> 4) & 31, ks = (t >> 9) & 3, b = t >> 11;
    int n = lane & 15;
    int k = ks * 32 + ((lane >> 4) << 4) + e;
    float v = (n < 8) ? Qf[((size_t)b * 8 + n) * 128 + k] : 0.f;
    qp[t] = (_Float16)v;
}

// ---------------------------------------------------------------------------
// Frontier extraction: stable compaction of channel-1 > 0, plus per-frontier
// agent mask byte (bit a = padding | dist_map[a] >= 4).
// ---------------------------------------------------------------------------
__global__ void __launch_bounds__(1024)
frontier_kernel(const float* __restrict__ inputs, int* __restrict__ rows,
                int* __restrict__ cols, int* __restrict__ counts,
                unsigned char* __restrict__ maskb) {
    int b = blockIdx.x;
    const float* ch1 = inputs + ((size_t)b * 16 + 1) * 65536;
    __shared__ int swv[32];
    __shared__ int sbase;
    int t = threadIdx.x, lane = t & 31, wv = t >> 5;
    if (t == 0) sbase = 0;
    __syncthreads();
    int* rb_ = rows + (size_t)b * MAXF;
    int* cb_ = cols + (size_t)b * MAXF;
    for (int chunk = 0; chunk < 65536; chunk += 1024) {
        int i = chunk + t;
        bool pred = ch1[i] > 0.f;
        unsigned int m32 = (unsigned int)__ballot(pred);
        int lp = __popc(m32 & ((1u << lane) - 1u));
        if (lane == 0) swv[wv] = __popc(m32);
        __syncthreads();
        if (t == 0) {
            int run = sbase;
            for (int w2 = 0; w2 < 32; ++w2) { int c = swv[w2]; swv[w2] = run; run += c; }
            sbase = run;
        }
        __syncthreads();
        if (pred) {
            int pos = swv[wv] + lp;
            if (pos < MAXF) { rb_[pos] = i >> 8; cb_[pos] = i & 255; }
        }
        __syncthreads();
    }
    int total = sbase;
    if (t == 0) counts[b] = total;
    for (int f = total + t; f < MAXF; f += 1024) { rb_[f] = 0; cb_[f] = 0; }
    __syncthreads();
    for (int f = t; f < MAXF; f += 1024) {
        unsigned char mk;
        if (f < total) {
            int r = rb_[f], c = cb_[f];
            mk = 0;
#pragma unroll
            for (int a = 0; a < 8; ++a) {
                float d = inputs[(((size_t)b * 16 + 8 + a) * 256 + r) * 256 + c];
                if (d >= 4.f) mk |= (unsigned char)(1u << a);
            }
        } else {
            mk = 0xFFu;
        }
        maskb[(size_t)b * MAXF + f] = mk;
    }
}

// ---------------------------------------------------------------------------
// Fused conv3x3 + BN + (residual) + (ReLU), implicit im2col on WMMA.
// Block = 256 thr = 8 waves; each wave = one 16-pixel output row strip,
// all COUT channels (NT accumulators). Patch staged to LDS as f16.
// ---------------------------------------------------------------------------
template <int CIN, int COUT, int S, int HIN, int WIN, bool RELU, bool RES>
__global__ void __launch_bounds__(256)
conv_bn_kernel(const float* __restrict__ in, const _Float16* __restrict__ wpack,
               const float* __restrict__ scale, const float* __restrict__ shift,
               const float* __restrict__ res, float* __restrict__ out) {
    constexpr int HO = HIN / S, WO = WIN / S;
    constexpr int TW = 16, TH = 8;
    constexpr int PH = (TH - 1) * S + 3, PW = (TW - 1) * S + 3;
    constexpr int KTOT = 9 * CIN;
    constexpr int KSTEPS = (KTOT + 31) / 32;
    constexpr int NT = COUT / 16;
    __shared__ _Float16 patch[PH * PW * CIN];

    int b = blockIdx.z;
    int ox0 = blockIdx.x * TW, oy0 = blockIdx.y * TH;
    const size_t inB = (size_t)b * CIN * HIN * WIN;

    for (int t = threadIdx.x; t < PH * PW * CIN; t += 256) {
        int cin = t % CIN, xy = t / CIN;
        int ix = xy % PW, iy = xy / PW;
        int gy = oy0 * S + iy - 1, gx = ox0 * S + ix - 1;
        float v = 0.f;
        if (gy >= 0 && gy < HIN && gx >= 0 && gx < WIN)
            v = in[inB + ((size_t)cin * HIN + gy) * WIN + gx];
        patch[t] = (_Float16)v;
    }
    __syncthreads();

    int wave = threadIdx.x >> 5, lane = threadIdx.x & 31;
    int wy = wave;                         // output row within tile
    v8f acc[NT];
#pragma unroll
    for (int n = 0; n < NT; ++n)
#pragma unroll
        for (int e = 0; e < 8; ++e) acc[n][e] = 0.f;

#pragma unroll
    for (int ks = 0; ks < KSTEPS; ++ks) {
        int m = lane & 15;                                // pixel x in tile
        int kb = ks * 32 + ((lane >> 4) << 4);
        v16h a;
#pragma unroll
        for (int e = 0; e < 16; ++e) {
            int k = kb + e;                               // {kb..+15}
            _Float16 v = (_Float16)0.f;
            if (k < KTOT) {
                int tap = k / CIN, cin = k % CIN;
                int dy = tap / 3, dx = tap % 3;
                v = patch[((wy * S + dy) * PW + (m * S + dx)) * CIN + cin];
            }
            a[e] = v;
        }
#pragma unroll
        for (int n = 0; n < NT; ++n) {
            v16h bf = load_b_frag(wpack + (size_t)((n * KSTEPS + ks) * 32) * 16);
            acc[n] = wmma_f16(a, bf, acc[n]);
        }
    }

    int oy = oy0 + wy;
    int col = lane & 15, mbase = (lane >> 4) << 3;
#pragma unroll
    for (int n = 0; n < NT; ++n) {
        int ch = n * 16 + col;
        float sc = scale[ch], sh = shift[ch];
#pragma unroll
        for (int r = 0; r < 8; ++r) {
            int ox = ox0 + mbase + r;
            size_t o = ((size_t)(b * COUT + ch) * HO + oy) * WO + ox;
            float v = acc[n][r] * sc + sh;
            if constexpr (RES) v += res[o];
            if constexpr (RELU) v = fmaxf(v, 0.f);
            out[o] = v;
        }
    }
}

// ---------------------------------------------------------------------------
// Agent-side: agent_full -> ae (66->128->128) -> Q (128->128), and
// gagent = mean_a mlp2(agent_full, 66->64->32). One block per batch.
// ---------------------------------------------------------------------------
__global__ void __launch_bounds__(128)
agent_kernel(const float* __restrict__ extras, const float* __restrict__ mf,
             const float* a1w, const float* a1b, const float* a2w, const float* a2b,
             const float* wqw, const float* wqb,
             const float* p1w, const float* p1b, const float* p2w, const float* p2b,
             float* __restrict__ Qout, float* __restrict__ gagent) {
    __shared__ float af[8][66];
    __shared__ float h1[8][128];
    __shared__ float ae_[8][128];
    __shared__ float g1[8][64];
    int b = blockIdx.x, t = threadIdx.x;

    for (int idx = t; idx < 8 * 66; idx += 128) {
        int a = idx / 66, i = idx % 66;
        float c0 = extras[b * 48 + a * 6 + 0];
        float c1 = extras[b * 48 + a * 6 + 1];
        float v;
        if (i < 64) {
            int ar = imin(imax((int)c0, 0), 63);
            int ac = imin(imax((int)c1, 0), 63);
            v = mf[((size_t)(b * 64 + i) * 64 + ar) * 64 + ac];
        } else {
            v = (i == 64 ? c0 : c1) * (1.f / 256.f);
        }
        af[a][i] = v;
    }
    __syncthreads();
    for (int a = 0; a < 8; ++a) {
        float s = a1b[t];
        for (int i = 0; i < 66; ++i) s += af[a][i] * a1w[i * 128 + t];
        h1[a][t] = fmaxf(s, 0.f);
    }
    __syncthreads();
    for (int a = 0; a < 8; ++a) {
        float s = a2b[t];
        for (int i = 0; i < 128; ++i) s += h1[a][i] * a2w[i * 128 + t];
        ae_[a][t] = s;
    }
    __syncthreads();
    for (int a = 0; a < 8; ++a) {
        float s = wqb[t];
        for (int i = 0; i < 128; ++i) s += ae_[a][i] * wqw[i * 128 + t];
        Qout[((size_t)b * 8 + a) * 128 + t] = s;
    }
    if (t < 64) {
        for (int a = 0; a < 8; ++a) {
            float s = p1b[t];
            for (int i = 0; i < 66; ++i) s += af[a][i] * p1w[i * 64 + t];
            g1[a][t] = fmaxf(s, 0.f);
        }
    }
    __syncthreads();
    if (t < 32) {
        float accm = 0.f;
        for (int a = 0; a < 8; ++a) {
            float s = p2b[t];
            for (int i = 0; i < 64; ++i) s += g1[a][i] * p2w[i * 32 + t];
            accm += s;
        }
        gagent[b * 32 + t] = accm * 0.125f;
    }
}

// ---------------------------------------------------------------------------
// Frontier side: fused (gather -> 64->128 relu -> 128->128 -> wk -> Q·K).
// 128 frontier rows per block, 8 waves, each wave owns a private 16-row slab
// of the LDS tile so the 4 GEMM stages need no barriers after staging.
// ---------------------------------------------------------------------------
template <int KSTEPS, int NT, bool RELU>
__device__ __forceinline__ void mlp_stage(_Float16* sH, int ld, int m0,
                                          const _Float16* wp, const float* bias) {
    int lane = threadIdx.x & 31;
    v16h afr[KSTEPS];
#pragma unroll
    for (int ks = 0; ks < KSTEPS; ++ks) afr[ks] = load_a_frag(sH, ld, m0, ks * 32);
#pragma unroll
    for (int nt = 0; nt < NT; ++nt) {
        v8f acc;
#pragma unroll
        for (int e = 0; e < 8; ++e) acc[e] = 0.f;
#pragma unroll
        for (int ks = 0; ks < KSTEPS; ++ks) {
            v16h bf = load_b_frag(wp + (size_t)((nt * KSTEPS + ks) * 32) * 16);
            acc = wmma_f16(afr[ks], bf, acc);
        }
        int n = nt * 16 + (lane & 15), mb = (lane >> 4) << 3;
        float bv = bias[n];
#pragma unroll
        for (int r = 0; r < 8; ++r) {
            float v = acc[r] + bv;
            if (RELU) v = fmaxf(v, 0.f);
            sH[(size_t)(m0 + mb + r) * ld + n] = (_Float16)v;
        }
    }
}

__global__ void __launch_bounds__(256)
frontier_attn_kernel(const float* __restrict__ mf, const int* __restrict__ rows,
                     const int* __restrict__ cols, const unsigned char* __restrict__ maskb,
                     const _Float16* __restrict__ w1p, const float* __restrict__ b1,
                     const _Float16* __restrict__ w2p, const float* __restrict__ b2,
                     const _Float16* __restrict__ wkp, const float* __restrict__ bk,
                     const _Float16* __restrict__ qpack, float* __restrict__ out_logits) {
    constexpr int LD = 136;                 // 128 + 8 pad (bank spread)
    __shared__ _Float16 sH[128 * LD];
    int b = blockIdx.y;
    int f0 = blockIdx.x * 128;
    int tid = threadIdx.x, lane = tid & 31, wave = tid >> 5;

    // stage gathered frontier map features (128 rows x 64 ch)
    for (int t = tid; t < 128 * 64; t += 256) {
        int c = t & 63, row = t >> 6;
        int f = f0 + row;
        int r = imin(rows[(size_t)b * MAXF + f], 63);
        int cc = imin(cols[(size_t)b * MAXF + f], 63);
        sH[(size_t)row * LD + c] = (_Float16)mf[((size_t)(b * 64 + c) * 64 + r) * 64 + cc];
    }
    __syncthreads();

    int m0 = wave * 16;
    mlp_stage<2, 8, true>(sH, LD, m0, w1p, b1);   // h1 = relu(x @ W1 + b1)
    mlp_stage<4, 8, false>(sH, LD, m0, w2p, b2);  // fe = h1 @ W2 + b2
    mlp_stage<4, 8, false>(sH, LD, m0, wkp, bk);  // K  = fe @ wk + bk

    // logits tile: K(16x128) @ Q_b^T(128x16-padded)
    v8f acc;
#pragma unroll
    for (int e = 0; e < 8; ++e) acc[e] = 0.f;
#pragma unroll
    for (int ks = 0; ks < 4; ++ks) {
        v16h a = load_a_frag(sH, LD, m0, ks * 32);
        v16h bf = load_b_frag(qpack + (size_t)((b * 4 + ks) * 32) * 16);
        acc = wmma_f16(a, bf, acc);
    }
    int a_idx = lane & 15, mb = (lane >> 4) << 3;
    if (a_idx < 8) {
        const float inv_scale = 0.04419417382415922f;   // 1/(NH*sqrt(HD))
#pragma unroll
        for (int r = 0; r < 8; ++r) {
            int f = f0 + m0 + mb + r;
            unsigned char mk = maskb[(size_t)b * MAXF + f];
            float v = ((mk >> a_idx) & 1) ? -1e9f : acc[r] * inv_scale;
            out_logits[((size_t)(b * 8 + a_idx)) * MAXF + f] = v;
        }
    }
}

// ---------------------------------------------------------------------------
// gmap = mean over 64x64 of relu(conv(mf)) per (b,c)
// ---------------------------------------------------------------------------
__global__ void __launch_bounds__(256)
gmap_kernel(const float* __restrict__ cr, float* __restrict__ gmap) {
    int bc = blockIdx.x;
    const float* p = cr + (size_t)bc * 4096;
    float s = 0.f;
    for (int i = threadIdx.x; i < 4096; i += 256) s += p[i];
    __shared__ float red[256];
    red[threadIdx.x] = s;
    __syncthreads();
    for (int st = 128; st > 0; st >>= 1) {
        if (threadIdx.x < st) red[threadIdx.x] += red[threadIdx.x + st];
        __syncthreads();
    }
    if (threadIdx.x == 0) gmap[bc] = red[0] * (1.f / 4096.f);
}

// ---------------------------------------------------------------------------
// Value head: concat(gmap,gagent) -> 64 relu -> 32 relu -> 1
// ---------------------------------------------------------------------------
__global__ void __launch_bounds__(64)
value_kernel(const float* __restrict__ gmap, const float* __restrict__ gagent,
             const float* v1w, const float* v1b, const float* v2w, const float* v2b,
             const float* v3w, const float* v3b, float* __restrict__ out) {
    __shared__ float comb[64], h1[64], h2[32];
    int b = blockIdx.x, t = threadIdx.x;
    comb[t] = (t < 32) ? gmap[b * 32 + t] : gagent[b * 32 + (t - 32)];
    __syncthreads();
    float a = v1b[t];
    for (int i = 0; i < 64; ++i) a += comb[i] * v1w[i * 64 + t];
    h1[t] = fmaxf(a, 0.f);
    __syncthreads();
    if (t < 32) {
        float a2 = v2b[t];
        for (int i = 0; i < 64; ++i) a2 += h1[i] * v2w[i * 32 + t];
        h2[t] = fmaxf(a2, 0.f);
    }
    __syncthreads();
    if (t == 0) {
        float a3 = v3b[0];
        for (int i = 0; i < 32; ++i) a3 += h2[i] * v3w[i];
        out[b] = a3;
    }
}

// ---------------------------------------------------------------------------
// Host orchestration
// ---------------------------------------------------------------------------
extern "C" void kernel_launch(void* const* d_in, const int* in_sizes, int n_in,
                              void* d_out, int out_size, void* d_ws, size_t ws_size,
                              hipStream_t stream) {
    (void)in_sizes; (void)n_in; (void)out_size; (void)ws_size;
    auto P = [&](int i) { return (const float*)d_in[i]; };
    const float* inputs = P(0);
    const float* extras = P(1);
    // param indices (dict insertion order)
    const float* e_c1_w = P(2);  const float* e_c1_b = P(3);
    const float* e_c2_w = P(8);  const float* e_c2_b = P(9);
    const float* rb_c1_w[3] = {P(14), P(26), P(38)};
    const float* rb_c1_b[3] = {P(15), P(27), P(39)};
    const float* rb_c2_w[3] = {P(20), P(32), P(44)};
    const float* rb_c2_b[3] = {P(21), P(33), P(45)};
    const float* a_e1_w = P(50); const float* a_e1_b = P(51);
    const float* a_e2_w = P(52); const float* a_e2_b = P(53);
    const float* f_e1_w = P(54); const float* f_e1_b = P(55);
    const float* f_e2_w = P(56); const float* f_e2_b = P(57);
    const float* wq_w = P(58);   const float* wq_b = P(59);
    const float* wk_w = P(60);   const float* wk_b = P(61);
    const float* cr_c_w = P(62); const float* cr_c_b = P(63);
    const float* cr_p1_w = P(64); const float* cr_p1_b = P(65);
    const float* cr_p2_w = P(66); const float* cr_p2_b = P(67);
    const float* v1_w = P(68); const float* v1_b = P(69);
    const float* v2_w = P(70); const float* v2_b = P(71);
    const float* v3_w = P(72); const float* v3_b = P(73);

    // workspace carve
    char* base = (char*)d_ws;
    size_t cur = 0;
    auto carve = [&](size_t bytes) -> void* {
        void* p = base + cur;
        cur = (cur + bytes + 255) & ~(size_t)255;
        return p;
    };
    float* sc[9]; float* sh[9];
    for (int i = 0; i < 9; ++i) { sc[i] = (float*)carve(64 * 4); sh[i] = (float*)carve(64 * 4); }
    _Float16* wc1 = (_Float16*)carve((size_t)2 * 5 * 512 * 2);
    _Float16* wc2 = (_Float16*)carve((size_t)4 * 9 * 512 * 2);
    _Float16* wrb[6];
    for (int i = 0; i < 6; ++i) wrb[i] = (_Float16*)carve((size_t)4 * 18 * 512 * 2);
    _Float16* wcr = (_Float16*)carve((size_t)2 * 18 * 512 * 2);
    _Float16* wf1 = (_Float16*)carve((size_t)8 * 2 * 512 * 2);
    _Float16* wf2 = (_Float16*)carve((size_t)8 * 4 * 512 * 2);
    _Float16* wwk = (_Float16*)carve((size_t)8 * 4 * 512 * 2);
    float* Qf = (float*)carve((size_t)32 * 8 * 128 * 4);
    _Float16* Qp = (_Float16*)carve((size_t)32 * 2048 * 2);
    float* gmap = (float*)carve(32 * 32 * 4);
    float* gagent = (float*)carve(32 * 32 * 4);
    int* counts = (int*)carve(32 * 4);
    int* rows = (int*)carve((size_t)32 * MAXF * 4);
    int* cols = (int*)carve((size_t)32 * MAXF * 4);
    unsigned char* maskb = (unsigned char*)carve((size_t)32 * MAXF);
    float* bufA = (float*)carve((size_t)32 * 32 * 128 * 128 * 4);  // 64MB (conv1 out / cr out)
    float* bufB = (float*)carve((size_t)32 * 64 * 64 * 64 * 4);    // 32MB
    float* bufC = (float*)carve((size_t)32 * 64 * 64 * 64 * 4);    // 32MB

    // --- fold BN + pack weights ---
    fold_bn_kernel<<<1, 64, 0, stream>>>(e_c1_b, P(4), P(5), P(6), P(7), sc[0], sh[0], 32);
    fold_bn_kernel<<<1, 64, 0, stream>>>(e_c2_b, P(10), P(11), P(12), P(13), sc[1], sh[1], 64);
    const int bnIdx[3][2] = {{16, 22}, {28, 34}, {40, 46}};
    for (int i = 0; i < 3; ++i) {
        int g1 = bnIdx[i][0], g2 = bnIdx[i][1];
        fold_bn_kernel<<<1, 64, 0, stream>>>(rb_c1_b[i], P(g1), P(g1 + 1), P(g1 + 2), P(g1 + 3),
                                             sc[2 + 2 * i], sh[2 + 2 * i], 64);
        fold_bn_kernel<<<1, 64, 0, stream>>>(rb_c2_b[i], P(g2), P(g2 + 1), P(g2 + 2), P(g2 + 3),
                                             sc[3 + 2 * i], sh[3 + 2 * i], 64);
    }
    ident_scale_kernel<<<1, 64, 0, stream>>>(cr_c_b, sc[8], sh[8], 32);

    pack_conv_w_kernel<<<20, 256, 0, stream>>>(e_c1_w, wc1, 16, 32, 5);
    pack_conv_w_kernel<<<72, 256, 0, stream>>>(e_c2_w, wc2, 32, 64, 9);
    for (int i = 0; i < 3; ++i) {
        pack_conv_w_kernel<<<144, 256, 0, stream>>>(rb_c1_w[i], wrb[2 * i], 64, 64, 18);
        pack_conv_w_kernel<<<144, 256, 0, stream>>>(rb_c2_w[i], wrb[2 * i + 1], 64, 64, 18);
    }
    pack_conv_w_kernel<<<72, 256, 0, stream>>>(cr_c_w, wcr, 64, 32, 18);
    pack_lin_w_kernel<<<32, 256, 0, stream>>>(f_e1_w, wf1, 64, 128, 2);
    pack_lin_w_kernel<<<64, 256, 0, stream>>>(f_e2_w, wf2, 128, 128, 4);
    pack_lin_w_kernel<<<64, 256, 0, stream>>>(wk_w, wwk, 128, 128, 4);

    // --- frontier extraction ---
    frontier_kernel<<<32, 1024, 0, stream>>>(inputs, rows, cols, counts, maskb);

    // --- encoder ---
    conv_bn_kernel<16, 32, 2, 256, 256, true, false>
        <<<dim3(8, 16, 32), 256, 0, stream>>>(inputs, wc1, sc[0], sh[0], nullptr, bufA);
    conv_bn_kernel<32, 64, 2, 128, 128, true, false>
        <<<dim3(4, 8, 32), 256, 0, stream>>>(bufA, wc2, sc[1], sh[1], nullptr, bufB);
    for (int i = 0; i < 3; ++i) {
        conv_bn_kernel<64, 64, 1, 64, 64, true, false>
            <<<dim3(4, 8, 32), 256, 0, stream>>>(bufB, wrb[2 * i], sc[2 + 2 * i], sh[2 + 2 * i],
                                                 nullptr, bufC);
        conv_bn_kernel<64, 64, 1, 64, 64, true, true>
            <<<dim3(4, 8, 32), 256, 0, stream>>>(bufC, wrb[2 * i + 1], sc[3 + 2 * i], sh[3 + 2 * i],
                                                 bufB, bufB);
    }
    // mf = bufB (B,64,64,64)

    // --- agent side + Q pack ---
    agent_kernel<<<32, 128, 0, stream>>>(extras, bufB, a_e1_w, a_e1_b, a_e2_w, a_e2_b,
                                         wq_w, wq_b, cr_p1_w, cr_p1_b, cr_p2_w, cr_p2_b,
                                         Qf, gagent);
    pack_q_kernel<<<256, 256, 0, stream>>>(Qf, Qp);

    // --- critic map branch ---
    conv_bn_kernel<64, 32, 1, 64, 64, true, false>
        <<<dim3(4, 8, 32), 256, 0, stream>>>(bufB, wcr, sc[8], sh[8], nullptr, bufA);
    gmap_kernel<<<32 * 32, 256, 0, stream>>>(bufA, gmap);

    // --- value head (out[0..31]) ---
    value_kernel<<<32, 64, 0, stream>>>(gmap, gagent, v1_w, v1_b, v2_w, v2_b, v3_w, v3_b,
                                        (float*)d_out);

    // --- frontier MLP + attention logits (out[32..]) ---
    frontier_attn_kernel<<<dim3(64, 32), 256, 0, stream>>>(
        bufB, rows, cols, maskb, wf1, f_e1_b, wf2, f_e2_b, wwk, wk_b, Qp,
        (float*)d_out + 32);
}
